// QiskitQuantumDQN_70377334112273
// MI455X (gfx1250) — compile-verified, matrix operand order checked by reading, not run on hardware
//
#include <hip/hip_runtime.h>
#include <hip/hip_bf16.h>
#include <math.h>

// ---------------------------------------------------------------------------
// MI455X (gfx1250, wave32) quantum-DQN forward pass.
//   convs          : f16 implicit-GEMM, v_wmma_f32_16x16x32_f16, 2-4 acc/wave
//   fc1/fc2        : TDM (tensor_load_to_lds) stages 16xK A tile in LDS,
//                    8 waves x NACC accumulators read A via ds_load_b128
//   quantum sim    : one wave32 per batch element, 8 amps/lane in registers
// ---------------------------------------------------------------------------

typedef __attribute__((ext_vector_type(16))) _Float16 v16h;
typedef __attribute__((ext_vector_type(8)))  _Float16 v8h;
typedef __attribute__((ext_vector_type(8)))  float    v8f;

typedef __attribute__((ext_vector_type(4))) unsigned int u32x4;
typedef __attribute__((ext_vector_type(8))) int          i32x8;
typedef __attribute__((ext_vector_type(4))) int          i32x4;

#define DEVINL __device__ __forceinline__

// A/B 16-bit fragment (16x32 / 32x16): per ISA each lane holds two contiguous
// 8-half runs: K = kb..kb+7 (elems 0..7) and kb+16..kb+23 (elems 8..15),
// kb = (lane>=16) ? 8 : 0.  Contiguous source -> two 16-byte vector loads
// (global_load_b128 or ds_load_b128 depending on address space).
DEVINL v16h load_frag_contig(const _Float16* p) {
  v8h lo = *(const v8h*)p;
  v8h hi = *(const v8h*)(p + 16);
  v16h r;
#pragma unroll
  for (int i = 0; i < 8; ++i) { r[i] = lo[i]; r[i + 8] = hi[i]; }
  return r;
}

// ---------------------------------------------------------------------------
// Tensor Data Mover: DMA a [nrows x rowlen] f16 tile (row stride = rowlen)
// from global into LDS. Descriptor per CDNA5 ISA
// section 8 (group0: count/lds_addr/global_addr/type=2; group1: data_size=2B,
// tensor dims, tile dims, dim0 stride). 2D tile -> groups 2/3 zero.
// ---------------------------------------------------------------------------
#if __has_builtin(__builtin_amdgcn_tensor_load_to_lds)
#define HAVE_TDM 1
DEVINL void tdm_load_2d_f16(void* lds_dst, const void* gsrc,
                            unsigned rowlen, unsigned nrows,
                            unsigned tensor_rows) {
  const unsigned long long ga = (unsigned long long)(uintptr_t)gsrc;
  u32x4 g0;
  g0[0] = 1u;                                            // count=1, user desc
  g0[1] = (unsigned)(uintptr_t)lds_dst;                  // LDS byte address
  g0[2] = (unsigned)(ga & 0xFFFFFFFFu);                  // global_addr[31:0]
  g0[3] = (unsigned)((ga >> 32) & 0x1FFFFFFu) | (2u << 30);  // [56:32] | type=2
  const unsigned long long s0 = rowlen;                  // dim0 stride (elems)
  i32x8 g1;
  g1[0] = (int)(1u << 16);                               // data_size = 2 bytes
  g1[1] = (int)((rowlen & 0xFFFFu) << 16);               // tensor_dim0[15:0]
  g1[2] = (int)(((rowlen >> 16) & 0xFFFFu) |             // tensor_dim0[31:16]
                ((tensor_rows & 0xFFFFu) << 16));        // tensor_dim1[15:0]
  g1[3] = (int)(((tensor_rows >> 16) & 0xFFFFu) |        // tensor_dim1[31:16]
                ((rowlen & 0xFFFFu) << 16));             // tile_dim0 = rowlen
  g1[4] = (int)(nrows & 0xFFFFu);                        // tile_dim1; tile_dim2=0
  g1[5] = (int)(s0 & 0xFFFFFFFFu);                       // dim0_stride[31:0]
  g1[6] = (int)((s0 >> 32) & 0xFFFFu);                   // dim0_stride[47:32]
  g1[7] = 0;
  i32x4 gz = {0, 0, 0, 0};
#if __clang_major__ >= 23
  i32x8 gz8 = {0, 0, 0, 0, 0, 0, 0, 0};
  __builtin_amdgcn_tensor_load_to_lds(g0, g1, gz, gz, gz8, 0);
#else
  __builtin_amdgcn_tensor_load_to_lds(g0, g1, gz, gz, 0);
#endif
}
#else
#define HAVE_TDM 0
#endif

// ---------------------------------------------------------------------------
// Implicit-GEMM conv: C[M=B*OH*OW][N=COUT] = im2col(in) x wt^T, relu, f16 out.
// One 16-row M-tile per wave; NT=COUT/16 accumulators so the gathered A
// fragment is reused across all N-tiles (2-4 WMMAs per im2col gather).
// ---------------------------------------------------------------------------
template <int CIN, int HIN, int WIN, int COUT, int KSZ, int STRIDE,
          int HOUT, int WOUT, typename TIN>
__global__ __launch_bounds__(256) void conv_wmma(
    const TIN* __restrict__ in, const _Float16* __restrict__ wt,
    const float* __restrict__ bias, _Float16* __restrict__ out, int M) {
  constexpr int K = CIN * KSZ * KSZ;
  constexpr int NT = COUT / 16;
  const int lane = threadIdx.x & 31;
  const int wid  = blockIdx.x * 8 + (threadIdx.x >> 5);
  if (wid >= M / 16) return;                    // wave-uniform
  const int m0 = wid * 16;

  const int row = m0 + (lane & 15);
  const int bb  = row / (HOUT * WOUT);
  const int rr  = row % (HOUT * WOUT);
  const int oy  = (rr / WOUT) * STRIDE;
  const int ox  = (rr % WOUT) * STRIDE;
  const TIN* __restrict__ inb = in + (size_t)bb * CIN * HIN * WIN;
  const int kb = (lane >> 4) * 8;
  const _Float16* __restrict__ wbase = wt + (size_t)(lane & 15) * K + kb;

  v8f acc[NT] = {};
#pragma unroll 2
  for (int k0 = 0; k0 < K; k0 += 32) {
    v16h afrag;
#pragma unroll
    for (int e = 0; e < 16; ++e) {
      const int k = k0 + kb + (e < 8 ? e : e + 8);
      const int ci = k / (KSZ * KSZ);
      const int krem = k % (KSZ * KSZ);
      const int ky = krem / KSZ, kx = krem % KSZ;
      afrag[e] = (_Float16)(float)inb[((size_t)ci * HIN + (oy + ky)) * WIN + (ox + kx)];
    }
#pragma unroll
    for (int t = 0; t < NT; ++t) {
      __builtin_prefetch(wbase + (size_t)t * 16 * K + k0 + 256, 0, 1);
      v16h bfrag = load_frag_contig(wbase + (size_t)t * 16 * K + k0);
      acc[t] = __builtin_amdgcn_wmma_f32_16x16x32_f16(false, afrag, false, bfrag,
                                                      (short)0, acc[t], false, false);
    }
  }

  const int mr = m0 + ((lane >> 4) << 3);
#pragma unroll
  for (int t = 0; t < NT; ++t) {
    const int n = t * 16 + (lane & 15);
    const float bv = bias[n];
#pragma unroll
    for (int v = 0; v < 8; ++v) {
      const int m = mr + v;
      const int b2 = m / (HOUT * WOUT);
      const int r2 = m % (HOUT * WOUT);
      float val = acc[t][v] + bv;
      val = val > 0.f ? val : 0.f;
      out[(((size_t)b2 * COUT + n) * HOUT + (r2 / WOUT)) * WOUT + (r2 % WOUT)] =
          (_Float16)val;
    }
  }
}

// ---------------------------------------------------------------------------
// Dense GEMM with LDS-staged A: one workgroup per 16-row M-tile.
// TDM DMAs the 16xK A tile into LDS (98KB for fc1 -- fits the 320KB WGP LDS),
// then 8 waves x NACC n-tiles consume it: A frags via ds_load_b128, B frags
// streamed from global with prefetch. Requires N == 8*NACC*16, K % 32 == 0.
// ---------------------------------------------------------------------------
template <int NACC, bool RELU>
__global__ __launch_bounds__(256) void gemm_wmma_lds(
    const _Float16* __restrict__ A, const _Float16* __restrict__ Bt,
    const float* __restrict__ bias, _Float16* __restrict__ C,
    int M, int N, int K) {
  extern __shared__ _Float16 sA[];              // [16][K]
  const int lane = threadIdx.x & 31;
  const int wv   = threadIdx.x >> 5;
  const int m0   = blockIdx.x * 16;

#if HAVE_TDM
  if (wv == 0) {
    tdm_load_2d_f16(sA, A + (size_t)m0 * K, (unsigned)K, 16u, (unsigned)M);
    __builtin_amdgcn_s_wait_tensorcnt(0);
  }
#else
  // Cooperative fallback: 256 threads copy 16*K halfs as b128 chunks.
  for (int i = threadIdx.x; i < 16 * K / 8; i += 256)
    *(v8h*)(sA + (size_t)i * 8) = *(const v8h*)(A + (size_t)m0 * K + (size_t)i * 8);
#endif
  __syncthreads();

  const int kb = (lane >> 4) * 8;
  const _Float16* arow = sA + (size_t)(lane & 15) * K + kb;
  const _Float16* __restrict__ brow =
      Bt + (size_t)(wv * NACC * 16 + (lane & 15)) * K + kb;

  v8f acc[NACC] = {};
  for (int k0 = 0; k0 < K; k0 += 32) {
    v16h afrag = load_frag_contig(arow + k0);   // ds_load_b128 x2
#pragma unroll
    for (int t = 0; t < NACC; ++t) {
      __builtin_prefetch(brow + (size_t)t * 16 * K + k0 + 256, 0, 1);
      v16h bfrag = load_frag_contig(brow + (size_t)t * 16 * K + k0);
      acc[t] = __builtin_amdgcn_wmma_f32_16x16x32_f16(false, afrag, false, bfrag,
                                                      (short)0, acc[t], false, false);
    }
  }

  const int mr = m0 + ((lane >> 4) << 3);
#pragma unroll
  for (int t = 0; t < NACC; ++t) {
    const int n = (wv * NACC + t) * 16 + (lane & 15);
    const float bv = bias[n];
#pragma unroll
    for (int v = 0; v < 8; ++v) {
      float val = acc[t][v] + bv;
      if (RELU) val = val > 0.f ? val : 0.f;
      C[(size_t)(mr + v) * N + n] = (_Float16)val;
    }
  }
}

// --------------------------- small helper kernels --------------------------
__global__ void cvt_f32_f16(const float* __restrict__ s, _Float16* __restrict__ d,
                            int n) {
  int i = blockIdx.x * blockDim.x + threadIdx.x;
  if (i < n) d[i] = (_Float16)s[i];
}

// fc3: [1024,128] f16 -> [1024,8] f32 with tanh  (N=8 < WMMA tile; VALU)
__global__ void fc_tanh8(const _Float16* __restrict__ A, const float* __restrict__ W,
                         const float* __restrict__ bias, float* __restrict__ out) {
  int i = blockIdx.x * blockDim.x + threadIdx.x;
  if (i >= 1024 * 8) return;
  const int b = i >> 3, q = i & 7;
  const _Float16* a = A + (size_t)b * 128;
  const float* w = W + q * 128;
  float s = bias[q];
#pragma unroll 8
  for (int k = 0; k < 128; ++k) s += (float)a[k] * w[k];
  out[i] = tanhf(s);
}

// ---------------------------------------------------------------------------
// 8-qubit statevector sim: 256 amplitudes = 8 regs x 32 lanes (one wave/batch).
// Qubit q<3 lives in the register index, q>=3 in the lane index (shfl_xor).
// ---------------------------------------------------------------------------
struct c32 { float x, y; };
DEVINL c32 cmul(c32 a, c32 b) { return {a.x * b.x - a.y * b.y, a.x * b.y + a.y * b.x}; }
DEVINL c32 cadd(c32 a, c32 b) { return {a.x + b.x, a.y + b.y}; }

DEVINL void gate1(c32 a[8], int lane, int q, c32 u00, c32 u01, c32 u10, c32 u11) {
  c32 p[8];
  if (q >= 3) {
    const int lm = 1 << (q - 3);
#pragma unroll
    for (int j = 0; j < 8; ++j) {
      p[j].x = __shfl_xor(a[j].x, lm, 32);
      p[j].y = __shfl_xor(a[j].y, lm, 32);
    }
    const int bit = (lane >> (q - 3)) & 1;
#pragma unroll
    for (int j = 0; j < 8; ++j)
      a[j] = bit ? cadd(cmul(u10, p[j]), cmul(u11, a[j]))
                 : cadd(cmul(u00, a[j]), cmul(u01, p[j]));
  } else {
    const int m = 1 << q;
#pragma unroll
    for (int j = 0; j < 8; ++j) p[j] = a[j ^ m];
#pragma unroll
    for (int j = 0; j < 8; ++j)
      a[j] = ((j >> q) & 1) ? cadd(cmul(u10, p[j]), cmul(u11, a[j]))
                            : cadd(cmul(u00, a[j]), cmul(u01, p[j]));
  }
}

DEVINL void gate_cx(c32 a[8], int lane, int c, int t) {
  c32 p[8];
  if (t >= 3) {
    const int lm = 1 << (t - 3);
#pragma unroll
    for (int j = 0; j < 8; ++j) {
      p[j].x = __shfl_xor(a[j].x, lm, 32);
      p[j].y = __shfl_xor(a[j].y, lm, 32);
    }
  } else {
    const int m = 1 << t;
#pragma unroll
    for (int j = 0; j < 8; ++j) p[j] = a[j ^ m];
  }
#pragma unroll
  for (int j = 0; j < 8; ++j) {
    const int g = (lane << 3) | j;
    if ((g >> c) & 1) a[j] = p[j];
  }
}

__global__ __launch_bounds__(256) void qsim(const float* __restrict__ qin,
                                            const float* __restrict__ qw,
                                            float* __restrict__ exps) {
  const int lane = threadIdx.x & 31;
  const int bidx = blockIdx.x * 8 + (threadIdx.x >> 5);  // 1024 waves exactly
  c32 a[8];
#pragma unroll
  for (int j = 0; j < 8; ++j) a[j] = {0.f, 0.f};
  if (lane == 0) a[0].x = 1.f;                           // |0...0>

  const float inv = 0.70710678118654752440f;
  const float* ang = qin + (size_t)bidx * 8;
#pragma unroll
  for (int q = 0; q < 8; ++q) {
    gate1(a, lane, q, {inv, 0.f}, {inv, 0.f}, {inv, 0.f}, {-inv, 0.f});   // H
    const float t = 0.5f * ang[q];
    const float ct = __cosf(t), st = __sinf(t);
    gate1(a, lane, q, {ct, 0.f}, {-st, 0.f}, {st, 0.f}, {ct, 0.f});       // RY(data)
  }
  int idx = 0;
#pragma unroll
  for (int L = 0; L < 3; ++L) {
#pragma unroll
    for (int i = 0; i < 7; ++i) gate_cx(a, lane, i, i + 1);
#pragma unroll
    for (int i = 0; i < 8; ++i) {
      const float tx = 0.5f * qw[idx++];
      const float cx_ = __cosf(tx), sx = __sinf(tx);
      gate1(a, lane, i, {cx_, 0.f}, {0.f, -sx}, {0.f, -sx}, {cx_, 0.f});  // RX
      const float ty = 0.5f * qw[idx++];
      const float cy = __cosf(ty), sy = __sinf(ty);
      gate1(a, lane, i, {cy, 0.f}, {-sy, 0.f}, {sy, 0.f}, {cy, 0.f});     // RY
      const float tz = 0.5f * qw[idx++];
      const float cz = __cosf(tz), sz = __sinf(tz);
      gate1(a, lane, i, {cz, -sz}, {0.f, 0.f}, {0.f, 0.f}, {cz, sz});     // RZ
    }
  }

  float pr[8];
#pragma unroll
  for (int j = 0; j < 8; ++j) pr[j] = a[j].x * a[j].x + a[j].y * a[j].y;
#pragma unroll
  for (int q = 0; q < 8; ++q) {
    float e = 0.f;
#pragma unroll
    for (int j = 0; j < 8; ++j) {
      const int g = (lane << 3) | j;
      e += ((g >> q) & 1) ? -pr[j] : pr[j];
    }
#pragma unroll
    for (int s = 16; s > 0; s >>= 1) e += __shfl_xor(e, s, 32);
    if (lane == 0) exps[(size_t)bidx * 8 + q] = e;
  }
}

// Post-quantum heads (K=8, N=6 below WMMA tile size -> VALU)
__global__ void head1(const float* __restrict__ exps, const float* __restrict__ W,
                      const float* __restrict__ bias, float* __restrict__ h) {
  int i = blockIdx.x * blockDim.x + threadIdx.x;
  if (i >= 1024 * 128) return;
  const int b = i >> 7, j = i & 127;
  const float* e = exps + b * 8;
  const float* w = W + j * 8;
  float s = bias[j];
#pragma unroll
  for (int k = 0; k < 8; ++k) s += e[k] * w[k];
  h[i] = fmaxf(s, 0.f);
}

__global__ void head2(const float* __restrict__ h, const float* __restrict__ W,
                      const float* __restrict__ bias, float* __restrict__ out) {
  int i = blockIdx.x * blockDim.x + threadIdx.x;
  if (i >= 1024 * 6) return;
  const int b = i / 6, aidx = i % 6;
  const float* hh = h + (size_t)b * 128;
  const float* w = W + aidx * 128;
  float s = bias[aidx];
#pragma unroll 8
  for (int k = 0; k < 128; ++k) s += hh[k] * w[k];
  out[i] = s;
}

// ---------------------------------------------------------------------------
extern "C" void kernel_launch(void* const* d_in, const int* in_sizes, int n_in,
                              void* d_out, int out_size, void* d_ws, size_t ws_size,
                              hipStream_t stream) {
  const float* x    = (const float*)d_in[0];
  const float* w1   = (const float*)d_in[1];
  const float* b1   = (const float*)d_in[2];
  const float* w2   = (const float*)d_in[3];
  const float* b2   = (const float*)d_in[4];
  const float* w3   = (const float*)d_in[5];
  const float* b3   = (const float*)d_in[6];
  const float* fw1  = (const float*)d_in[7];
  const float* fb1  = (const float*)d_in[8];
  const float* fw2  = (const float*)d_in[9];
  const float* fb2  = (const float*)d_in[10];
  const float* fw3  = (const float*)d_in[11];
  const float* fb3  = (const float*)d_in[12];
  const float* qw   = (const float*)d_in[13];
  const float* qqw1 = (const float*)d_in[14];
  const float* qqb1 = (const float*)d_in[15];
  const float* qqw2 = (const float*)d_in[16];
  const float* qqb2 = (const float*)d_in[17];
  (void)in_sizes; (void)n_in; (void)out_size; (void)ws_size;

  char* ws = (char*)d_ws;
  size_t off = 0;
  auto alloc = [&](size_t bytes) {
    void* p = ws + off;
    off = (off + bytes + 255) & ~(size_t)255;
    return p;
  };
  _Float16* w1h  = (_Float16*)alloc((size_t)32 * 256 * 2);
  _Float16* w2h  = (_Float16*)alloc((size_t)64 * 512 * 2);
  _Float16* w3h  = (_Float16*)alloc((size_t)64 * 576 * 2);
  _Float16* fw1h = (_Float16*)alloc((size_t)512 * 3136 * 2);
  _Float16* fw2h = (_Float16*)alloc((size_t)128 * 512 * 2);
  _Float16* h1   = (_Float16*)alloc((size_t)1024 * 32 * 400 * 2);
  _Float16* h2   = (_Float16*)alloc((size_t)1024 * 64 * 81 * 2);
  _Float16* h3   = (_Float16*)alloc((size_t)1024 * 3136 * 2);
  _Float16* f1   = (_Float16*)alloc((size_t)1024 * 512 * 2);
  _Float16* f2   = (_Float16*)alloc((size_t)1024 * 128 * 2);
  float*    qin  = (float*)alloc((size_t)1024 * 8 * 4);
  float*    exps = (float*)alloc((size_t)1024 * 8 * 4);
  float*    h4   = (float*)alloc((size_t)1024 * 128 * 4);

  auto cvt = [&](const float* s, _Float16* d, int n) {
    cvt_f32_f16<<<(n + 255) / 256, 256, 0, stream>>>(s, d, n);
  };
  cvt(w1,  w1h,  32 * 256);
  cvt(w2,  w2h,  64 * 512);
  cvt(w3,  w3h,  64 * 576);
  cvt(fw1, fw1h, 512 * 3136);
  cvt(fw2, fw2h, 128 * 512);

  // conv1: [1024,4,84,84] -> [1024,32,20,20]   M=409600 N=32 K=256 (2 acc/wave)
  conv_wmma<4, 84, 84, 32, 8, 4, 20, 20, float>
      <<<409600 / 16 / 8, 256, 0, stream>>>(x, w1h, b1, h1, 409600);
  // conv2: -> [1024,64,9,9]                    M=82944 N=64 K=512 (4 acc/wave)
  conv_wmma<32, 20, 20, 64, 4, 2, 9, 9, _Float16>
      <<<82944 / 16 / 8, 256, 0, stream>>>(h1, w2h, b2, h2, 82944);
  // conv3: -> [1024,64,7,7]                    M=50176 N=64 K=576 (4 acc/wave)
  conv_wmma<64, 9, 9, 64, 3, 1, 7, 7, _Float16>
      <<<50176 / 16 / 8, 256, 0, stream>>>(h2, w3h, b3, h3, 50176);

  // fc1: [1024,3136] x [512,3136]^T, TDM-staged A tile (98KB LDS), N=8*4*16
  gemm_wmma_lds<4, true><<<1024 / 16, 256, 16 * 3136 * 2, stream>>>(
      h3, fw1h, fb1, f1, 1024, 512, 3136);
  // fc2: [1024,512] x [128,512]^T, TDM-staged A tile (16KB LDS), N=8*1*16
  gemm_wmma_lds<1, true><<<1024 / 16, 256, 16 * 512 * 2, stream>>>(
      f1, fw2h, fb2, f2, 1024, 128, 512);

  // fc3 + tanh -> qubit angles [1024,8] f32
  fc_tanh8<<<(1024 * 8 + 255) / 256, 256, 0, stream>>>(f2, fw3, fb3, qin);

  // 8-qubit exact statevector sim, one wave32 per batch element
  qsim<<<1024 / 8, 256, 0, stream>>>(qin, qw, exps);

  // small heads
  head1<<<(1024 * 128 + 255) / 256, 256, 0, stream>>>(exps, qqw1, qqb1, h4);
  head2<<<(1024 * 6 + 255) / 256, 256, 0, stream>>>(h4, qqw2, qqb2, (float*)d_out);
}
